// TopologyAwarePatchSelector_35373350650266
// MI455X (gfx1250) — matrix operand order repaired and checked by reading, not compiled, not run on hardware
//
#include <hip/hip_runtime.h>
#include <hip/hip_bf16.h>

typedef __attribute__((ext_vector_type(16))) __bf16 v16bf;
typedef __attribute__((ext_vector_type(8)))  float  v8f;

#define HD        192
#define TWO_H     384
#define PPAT      64
#define RADIUS_C  0.42f
#define BSCALE_C  0.18f
#define NEG_INF_C (-1e9f)
#define WAVES     4
#define SCRS      66      // 16x66 scratch stride: conflict-free for half-split reads

__device__ __forceinline__ v8f wmma_bf16(v16bf a, v16bf b, v8f c) {
  return __builtin_amdgcn_wmma_f32_16x16x32_bf16(false, a, false, b, (short)0, c, false, false);
}

__global__ __launch_bounds__(128)
__attribute__((amdgpu_waves_per_eu(2)))
void topo_patch_kernel(const float* __restrict__ ts, const float* __restrict__ ps,
                       const float* __restrict__ Wsel, const float* __restrict__ bsel,
                       const float* __restrict__ Wbr,  const float* __restrict__ bbr,
                       const float* __restrict__ pv,   const float* __restrict__ ppos,
                       float* __restrict__ out, int Bn)
{
  // LDS: parameters pre-swizzled into WMMA bf16 B-fragment order.
  __shared__ __bf16 sWsel[48*32*16];   // 12 ksteps x 4 ntiles
  __shared__ __bf16 sWbr [48*32*16];
  __shared__ __bf16 sPV  [24*32*16];   // 2 ksteps x 12 ntiles
  __shared__ float  sPos[PPAT*3];
  __shared__ float  sBs[PPAT];
  __shared__ float  sBb[PPAT];
  __shared__ float  sScr[WAVES][2][16*SCRS];

  // ---- stage parameters (f32 -> bf16, fragment order) ----
  for (int idx = threadIdx.x; idx < 48*32*16; idx += blockDim.x) {
    const int e  = idx & 15;
    const int lf = (idx >> 4) & 31;
    const int f  = idx >> 9;
    const int kst = f >> 2, nt = f & 3;
    const int k = kst*32 + (lf >> 4)*8 + ((e < 8) ? e : (e + 8));   // B (KxN): lanes 0-15 K 0-7,16-23; 16-31 K 8-15,24-31
    const int n = nt*16 + (lf & 15);
    sWsel[idx] = (__bf16)Wsel[n*TWO_H + k];    // B(k,n) = W[n][k]  (logits = q @ W^T)
    sWbr [idx] = (__bf16)Wbr [n*TWO_H + k];
  }
  for (int idx = threadIdx.x; idx < 24*32*16; idx += blockDim.x) {
    const int e  = idx & 15;
    const int lf = (idx >> 4) & 31;
    const int f  = idx >> 9;
    const int kst = f / 12, nt = f % 12;
    const int k = kst*32 + (lf >> 4)*8 + ((e < 8) ? e : (e + 8));
    const int n = nt*16 + (lf & 15);
    sPV[idx] = (__bf16)pv[k*HD + n];           // B(k,n) = patch_values[k][n]
  }
  if (threadIdx.x < PPAT) {
    sBs[threadIdx.x] = bsel[threadIdx.x];
    sBb[threadIdx.x] = bbr[threadIdx.x];
    sPos[threadIdx.x*3+0] = ppos[threadIdx.x*3+0];
    sPos[threadIdx.x*3+1] = ppos[threadIdx.x*3+1];
    sPos[threadIdx.x*3+2] = ppos[threadIdx.x*3+2];
  }
  __syncthreads();

  const int lane  = threadIdx.x & 31;
  const int wv    = threadIdx.x >> 5;
  const int rhalf = lane >> 4;     // 0: lanes 0-15, 1: lanes 16-31
  const int l15   = lane & 15;
  const int kbase = rhalf * 32;    // epilogue: this lane owns patches [kbase, kbase+32)

  const size_t oW  = (size_t)Bn * HD;
  const size_t oA  = oW  + (size_t)Bn * PPAT;
  const size_t oLM = oA  + (size_t)Bn * 3;
  const size_t oBM = oLM + (size_t)Bn;
  const size_t oMD = oBM + (size_t)Bn;
  const size_t oR  = oMD + (size_t)Bn;
  const size_t oBS = oR  + (size_t)Bn;

  float* bufA = &sScr[wv][0][0];
  float* bufB = &sScr[wv][1][0];

  const int nTiles = Bn >> 4;
  const int GW  = gridDim.x * WAVES;
  const int gw  = blockIdx.x * WAVES + wv;
  const int tpw = (nTiles + GW - 1) / GW;

  for (int it = 0; it < tpw; ++it) {
    const int tile = gw + it * GW;
    const bool active = tile < nTiles;       // wave-uniform
    const int row_g = tile*16 + l15;

    v8f accS[4], accB[4];
    #pragma unroll
    for (int nt = 0; nt < 4; ++nt) {
      accS[nt] = (v8f){0.f,0.f,0.f,0.f,0.f,0.f,0.f,0.f};
      accB[nt] = (v8f){0.f,0.f,0.f,0.f,0.f,0.f,0.f,0.f};
    }

    // ---- logits GEMM: 16x64 = [16x384] x [384x64], bf16 WMMA, K in 12 steps ----
    if (active) {
      const int ntile = tile + GW;
      if (ntile < nTiles) {
        const size_t nro = (size_t)(ntile*16 + l15) * HD + rhalf*96;
        __builtin_prefetch(ts + nro, 0, 1);
        __builtin_prefetch(ps + nro, 0, 1);
      }
      #pragma unroll 6
      for (int kst = 0; kst < 12; ++kst) {
        const float* src = (kst < 6 ? ts : ps)
                         + (size_t)row_g * HD + (kst < 6 ? kst*32 : kst*32 - 192)
                         + rhalf*8;
        float qv[16];
        *(float4*)&qv[0]  = *(const float4*)(src +  0);
        *(float4*)&qv[4]  = *(const float4*)(src +  4);
        *(float4*)&qv[8]  = *(const float4*)(src + 16);
        *(float4*)&qv[12] = *(const float4*)(src + 20);
        v16bf a;
        #pragma unroll
        for (int e = 0; e < 16; ++e) a[e] = (__bf16)qv[e];
        #pragma unroll
        for (int nt = 0; nt < 4; ++nt) {
          const int f = kst*4 + nt;
          v16bf bS = *(const v16bf*)&sWsel[(f*32 + lane)*16];
          v16bf bB = *(const v16bf*)&sWbr [(f*32 + lane)*16];
          accS[nt] = wmma_bf16(a, bS, accS[nt]);
          accB[nt] = wmma_bf16(a, bB, accB[nt]);
        }
      }
    }

    // ---- spill C fragments (+bias) to per-wave LDS scratch ----
    #pragma unroll
    for (int nt = 0; nt < 4; ++nt) {
      const int col = nt*16 + l15;
      const float vbS = sBs[col], vbB = sBb[col];
      #pragma unroll
      for (int r = 0; r < 8; ++r) {
        const int rowl = rhalf*8 + r;
        bufA[rowl*SCRS + col] = accS[nt][r] + vbS;
        bufB[rowl*SCRS + col] = accB[nt][r] + vbB;
      }
    }
    __syncthreads();

    // ================= per-row epilogue =================
    // lane (l15, rhalf) owns row l15, patches [kbase, kbase+32); halves merge via shfl_xor(16)
    float* ba = bufA + l15*SCRS + kbase;   // base logits -> exp -> final weights
    float* bb = bufB + l15*SCRS + kbase;   // bridge logits -> mixed
    const float* pos = sPos + kbase*3;

    float mx = -3.4e38f;
    #pragma clang loop unroll_count(8)
    for (int j = 0; j < 32; ++j) mx = fmaxf(mx, ba[j]);
    mx = fmaxf(mx, __shfl_xor(mx, 16));

    float se = 0.f, ax = 0.f, ay = 0.f, az = 0.f;
    #pragma clang loop unroll_count(4)
    for (int j = 0; j < 32; ++j) {
      const float e = __expf(ba[j] - mx);
      se += e;
      ax += e * pos[j*3+0]; ay += e * pos[j*3+1]; az += e * pos[j*3+2];
      ba[j] = e;
    }
    se += __shfl_xor(se, 16);
    ax += __shfl_xor(ax, 16);
    ay += __shfl_xor(ay, 16);
    az += __shfl_xor(az, 16);
    const float inv_se = 1.f / se;
    ax *= inv_se; ay *= inv_se; az *= inv_se;

    // argmin of distance (strict < keeps lowest index within half)
    float bd = 3.4e38f; int ni = 1 << 30;
    #pragma clang loop unroll_count(4)
    for (int j = 0; j < 32; ++j) {
      const float dx = ax - pos[j*3+0], dy = ay - pos[j*3+1], dz = az - pos[j*3+2];
      const float d2 = dx*dx + dy*dy + dz*dz;
      if (d2 < bd) { bd = d2; ni = kbase + j; }
    }
    {
      const float obd = __shfl_xor(bd, 16);
      const int   oni = __shfl_xor(ni, 16);
      if (obd < bd || (obd == bd && oni < ni)) { bd = obd; ni = oni; }
    }

    // local mask + local top-6 of masked bridge logits over this half
    unsigned lmask = 0u;
    float lv[6]; int li[6];
    #pragma unroll
    for (int i = 0; i < 6; ++i) { lv[i] = NEG_INF_C; li[i] = 1 << 30; }
    #pragma clang loop unroll(disable)
    for (int j = 0; j < 32; ++j) {
      const float dx = ax - pos[j*3+0], dy = ay - pos[j*3+1], dz = az - pos[j*3+2];
      const float d = sqrtf(dx*dx + dy*dy + dz*dz);
      const bool lm = (d <= RADIUS_C) || ((kbase + j) == ni);
      lmask |= lm ? (1u << j) : 0u;
      const float m = lm ? NEG_INF_C : bb[j];
      if (m > lv[5]) {
        lv[5] = m; li[5] = kbase + j;
        #pragma unroll
        for (int q = 5; q > 0; --q)
          if (lv[q] > lv[q-1] || (lv[q] == lv[q-1] && li[q] < li[q-1])) {
            float t = lv[q]; lv[q] = lv[q-1]; lv[q-1] = t;
            int   u = li[q]; li[q] = li[q-1]; li[q-1] = u;
          }
      }
    }

    // merge the two sorted 6-lists: pad to 8 desc + partner reversed asc -> bitonic 16
    float mv[16]; int mi[16];
    #pragma unroll
    for (int i = 0; i < 6; ++i) { mv[i] = lv[i]; mi[i] = li[i]; }
    mv[6] = NEG_INF_C; mi[6] = 1 << 30;
    mv[7] = NEG_INF_C; mi[7] = 1 << 30;
    #pragma unroll
    for (int i = 0; i < 8; ++i) {
      const int src = 7 - i;
      mv[8+i] = (src < 6) ? __shfl_xor(lv[src], 16) : NEG_INF_C;
      mi[8+i] = (src < 6) ? __shfl_xor(li[src], 16) : (1 << 30);
    }
    #pragma unroll
    for (int d = 8; d >= 1; d >>= 1) {
      #pragma unroll
      for (int i = 0; i < 16; ++i) {
        if (!(i & d)) {
          const int j2 = i | d;
          const bool sw = (mv[i] < mv[j2]) || (mv[i] == mv[j2] && mi[i] > mi[j2]);
          if (sw) {
            float t = mv[i]; mv[i] = mv[j2]; mv[j2] = t;
            int   u = mi[i]; mi[i] = mi[j2]; mi[j2] = u;
          }
        }
      }
    }
    // top-6 softmax (both halves hold identical merged list)
    float tsum = 0.f, wsm[6];
    #pragma unroll
    for (int i = 0; i < 6; ++i) { wsm[i] = __expf(mv[i] - mv[0]); tsum += wsm[i]; }
    const float inv_ts = 1.f / tsum;

    const float inv_2r2 = 1.f / (2.f * RADIUS_C * RADIUS_C);
    float total = 0.f;
    #pragma clang loop unroll_count(2)
    for (int j = 0; j < 32; ++j) {
      const int k = kbase + j;
      const float dx = ax - pos[j*3+0], dy = ay - pos[j*3+1], dz = az - pos[j*3+2];
      const float d2 = dx*dx + dy*dy + dz*dz;
      const float kern = __expf(-d2 * inv_2r2);
      const bool  lm   = (lmask >> j) & 1u;
      float sp = 0.f;
      #pragma unroll
      for (int i = 0; i < 6; ++i) sp = (k == mi[i]) ? wsm[i]*inv_ts : sp;
      const float mixed = (ba[j]*inv_se) * kern * (lm ? 1.f : 0.f) + BSCALE_C * sp;
      total += mixed;
      bb[j] = mixed;
    }
    total += __shfl_xor(total, 16);
    const float inv_t = 1.f / fmaxf(total, 1e-6f);

    float lmass = 0.f, bmass = 0.f, md = 0.f;
    float* wout = out + oW + (size_t)row_g*PPAT + kbase;
    #pragma clang loop unroll(disable)
    for (int g = 0; g < 8; ++g) {
      float tmp[4];
      #pragma unroll
      for (int q = 0; q < 4; ++q) {
        const int j = g*4 + q;
        const float w = bb[j] * inv_t;
        ba[j] = w;                                // final weights, reused for last GEMM
        const float dx = ax - pos[j*3+0], dy = ay - pos[j*3+1], dz = az - pos[j*3+2];
        const float d = sqrtf(dx*dx + dy*dy + dz*dz);
        const bool lm = (lmask >> j) & 1u;
        lmass += lm ? w : 0.f;
        bmass += lm ? 0.f : w;
        md    += w * d;
        tmp[q] = w;
      }
      if (active) *(float4*)(wout + g*4) = make_float4(tmp[0], tmp[1], tmp[2], tmp[3]);
    }
    lmass += __shfl_xor(lmass, 16);
    bmass += __shfl_xor(bmass, 16);
    md    += __shfl_xor(md, 16);
    if (active && lane < 16) {
      const size_t rg = (size_t)row_g;
      out[oA + rg*3 + 0] = ax; out[oA + rg*3 + 1] = ay; out[oA + rg*3 + 2] = az;
      out[oLM + rg] = lmass;
      out[oBM + rg] = bmass;
      out[oMD + rg] = md;
      out[oR  + rg] = RADIUS_C;
      out[oBS + rg] = BSCALE_C;
    }
    __syncthreads();

    // ---- patch_state = weights[16x64] @ patch_values[64x192], bf16 WMMA ----
    if (active) {
      v16bf aw[2];
      #pragma unroll
      for (int kst = 0; kst < 2; ++kst)
        #pragma unroll
        for (int e = 0; e < 16; ++e) {
          const int kk = kst*32 + rhalf*8 + ((e < 8) ? e : (e + 8));
          aw[kst][e] = (__bf16)bufA[l15*SCRS + kk];
        }
      #pragma unroll
      for (int nt = 0; nt < 12; ++nt) {
        v8f acc = (v8f){0.f,0.f,0.f,0.f,0.f,0.f,0.f,0.f};
        v16bf b0 = *(const v16bf*)&sPV[(( 0 + nt)*32 + lane)*16];
        v16bf b1 = *(const v16bf*)&sPV[((12 + nt)*32 + lane)*16];
        acc = wmma_bf16(aw[0], b0, acc);
        acc = wmma_bf16(aw[1], b1, acc);
        #pragma unroll
        for (int r = 0; r < 8; ++r) {
          const int rowl = rhalf*8 + r;
          out[(size_t)(tile*16 + rowl)*HD + nt*16 + l15] = acc[r];
        }
      }
    }
    __syncthreads();
  }
}

extern "C" void kernel_launch(void* const* d_in, const int* in_sizes, int n_in,
                              void* d_out, int out_size, void* d_ws, size_t ws_size,
                              hipStream_t stream) {
  const float* ts   = (const float*)d_in[0];
  const float* ps   = (const float*)d_in[1];
  const float* Wsel = (const float*)d_in[2];
  const float* bs   = (const float*)d_in[3];
  const float* Wbr  = (const float*)d_in[4];
  const float* bb   = (const float*)d_in[5];
  const float* pv   = (const float*)d_in[6];
  const float* pp   = (const float*)d_in[7];
  float* out = (float*)d_out;
  const int Bn = in_sizes[0] / HD;
  topo_patch_kernel<<<dim3(1024), dim3(128), 0, stream>>>(ts, ps, Wsel, bs, Wbr, bb, pv, pp, out, Bn);
}
